// SpatialMemoryModule_32014686224741
// MI455X (gfx1250) — matrix operand order repaired and checked by reading, not compiled, not run on hardware
//
#include <hip/hip_runtime.h>

// ---------------------------------------------------------------------------
// SpatialMemoryModule fused attention for gfx1250 (MI455X), bf16 WMMA path.
//   q = x @ Wq^T ; k,v = mem @ W^T ; ssim(q,k) ; softmax ; out = att @ v
// ---------------------------------------------------------------------------

#define HW    256
#define MEMD  1024
#define C1f   0.01f
#define C2f   0.03f
#define EPSf  1e-8f
#define QSTR  264   // LDS row stride (bf16) for q strip: 256 + 8 pad -> conflict-free b128 reads
#define PSTR  40    // LDS row stride (bf16) for p tile:   32 + 8 pad

typedef __attribute__((ext_vector_type(16))) __bf16 v16bf;
typedef __attribute__((ext_vector_type(8)))  float  v8f;

typedef uint4 __attribute__((may_alias)) uint4a;
typedef uint2 __attribute__((may_alias)) uint2a;

__device__ inline v8f wmma_bf16(v16bf a, v16bf b, v8f c) {
  // D = A(16x32 bf16) x B(32x16 bf16) + C(16x16 f32)
  return __builtin_amdgcn_wmma_f32_16x16x32_bf16(false, a, false, b, (short)0, c,
                                                 false, false);
}

// B fragment: 16 contiguous bf16 starting at p (lane supplies its own row/K base).
__device__ inline v16bf ld16(const __bf16* p) {
  union { uint4a u[2]; v16bf v; } f;
  f.u[0] = *(const uint4a*)(p);
  f.u[1] = *(const uint4a*)(p + 8);
  return f.v;
}

// A fragment: two 8-element (16 B) chunks (K and K+8: 16 contiguous k).
__device__ inline v16bf ld8x2(const __bf16* p0, const __bf16* p1) {
  union { uint4a u[2]; v16bf v; } f;
  f.u[0] = *(const uint4a*)(p0);
  f.u[1] = *(const uint4a*)(p1);
  return f.v;
}

__device__ inline unsigned pack_bf16(float a, float b) {
  __bf16 x = (__bf16)a, y = (__bf16)b;
  unsigned short ux = __builtin_bit_cast(unsigned short, x);
  unsigned short uy = __builtin_bit_cast(unsigned short, y);
  return (unsigned)ux | ((unsigned)uy << 16);
}

// ---------------------------------------------------------------------------
// Prep: Wq -> bf16
// ---------------------------------------------------------------------------
__global__ void smm_wconv(const float* __restrict__ Wq, __bf16* __restrict__ Wqb) {
  int i = blockIdx.x * 256 + threadIdx.x;
  Wqb[i] = (__bf16)Wq[i];
}

// ---------------------------------------------------------------------------
// Prep: k = mem@Wk^T, v = mem@Wv^T ; k row stats (ddof=1) ; store centered-k
// bf16 [1024][256] and v^T bf16 [256][1024]. One block per memory slot.
// ---------------------------------------------------------------------------
__global__ __launch_bounds__(256) void smm_prep(
    const float* __restrict__ mem, const float* __restrict__ Wk,
    const float* __restrict__ Wv, __bf16* __restrict__ kcb,
    __bf16* __restrict__ vTb, float* __restrict__ kmw, float* __restrict__ kvw) {
  __shared__ float mrow[HW];
  __shared__ float red[256];
  __shared__ float red2[256];
  const int t = threadIdx.x;
  const int j = blockIdx.x;
  mrow[t] = mem[j * HW + t];
  __syncthreads();
  const float* wk = Wk + t * HW;
  const float* wv = Wv + t * HW;
  float ak = 0.f, av = 0.f;
#pragma unroll 8
  for (int c = 0; c < HW; ++c) {
    float m = mrow[c];
    ak += m * wk[c];
    av += m * wv[c];
  }
  red[t]  = ak;
  red2[t] = ak * ak;
  __syncthreads();
  for (int s = 128; s > 0; s >>= 1) {
    if (t < s) { red[t] += red[t + s]; red2[t] += red2[t + s]; }
    __syncthreads();
  }
  const float mu  = red[0] * (1.0f / HW);
  const float var = (red2[0] - (float)HW * mu * mu) * (1.0f / (HW - 1));
  kcb[j * HW + t] = (__bf16)(ak - mu);   // centered k, row-major [MEMD][HW]
  vTb[t * MEMD + j] = (__bf16)av;        // v transposed [HW][MEMD]
  if (t == 0) { kmw[j] = mu; kvw[j] = var; }
}

// ---------------------------------------------------------------------------
// Main fused kernel: 8 waves/block, 16 token rows per wave, 512 blocks.
// ---------------------------------------------------------------------------
__global__ __launch_bounds__(256) void smm_main(
    const float* __restrict__ x, const __bf16* __restrict__ Wqb,
    const __bf16* __restrict__ kcb, const __bf16* __restrict__ vTb,
    const float* __restrict__ kmw, const float* __restrict__ kvw,
    float* __restrict__ out) {
  __shared__ __align__(16) __bf16 qbuf[8][16][QSTR];
  __shared__ __align__(16) __bf16 pbuf[8][16][PSTR];

  const int tid  = threadIdx.x;
  const int wave = tid >> 5;
  const int lane = tid & 31;
  const int lrow = lane & 15;  // row (A) / column (B,C) index within tile
  const int half = lane >> 4;  // lane-group selector
  const long r0  = (long)blockIdx.x * 128 + (long)wave * 16;

  __bf16(*qs)[QSTR] = qbuf[wave];
  __bf16(*ps)[PSTR] = pbuf[wave];

  // ---- Phase 1: load x strip [16][256] f32, convert to bf16 into LDS ----
#pragma unroll
  for (int c = 0; c < 32; ++c) {
    const int idx = c * 32 + lane;  // float4 index within strip (coalesced)
    const int row = idx >> 6;       // 64 float4 per row
    const int c4  = idx & 63;
    const float4 xv = *(const float4*)(x + (r0 + row) * HW + c4 * 4);
    uint2a* dst = (uint2a*)&qs[row][c4 * 4];
    *dst = make_uint2(pack_bf16(xv.x, xv.y), pack_bf16(xv.z, xv.w));
  }

  // ---- Phase 2: q = x @ Wq^T  (16x256 result in 16 C-tiles) ----
  v8f qacc[16];
#pragma unroll
  for (int ot = 0; ot < 16; ++ot) { v8f z = {}; qacc[ot] = z; }
#pragma unroll
  for (int kk = 0; kk < 8; ++kk) {
    const int k0 = kk * 32 + half * 16;
    const v16bf a = ld8x2(&qs[lrow][k0], &qs[lrow][k0 + 8]);
    const int kb = kk * 32 + half * 16;
#pragma unroll
    for (int ot = 0; ot < 16; ++ot) {
      const v16bf b = ld16(Wqb + (ot * 16 + lrow) * HW + kb);
      qacc[ot] = wmma_bf16(a, b, qacc[ot]);
    }
  }

  // ---- Phase 3: per-row mean / var(ddof=1). C-layout: VGPR g, lane L holds
  // row (g + 8*(L>>4)); 4 xor-shuffles reduce across the 16-lane half. ----
  float rsum[8] = {}, rsq[8] = {};
#pragma unroll
  for (int ot = 0; ot < 16; ++ot)
#pragma unroll
    for (int g = 0; g < 8; ++g) {
      const float v = qacc[ot][g];
      rsum[g] += v;
      rsq[g]  += v * v;
    }
  float qm[8], qv[8];
#pragma unroll
  for (int g = 0; g < 8; ++g) {
    float s = rsum[g], s2 = rsq[g];
#pragma unroll
    for (int m = 1; m <= 8; m <<= 1) {
      s  += __shfl_xor(s, m, 32);
      s2 += __shfl_xor(s2, m, 32);
    }
    const float mu = s * (1.0f / HW);
    qm[g] = mu;
    qv[g] = (s2 - (float)HW * mu * mu) * (1.0f / (HW - 1));
  }

  // ---- Phase 4: center q, store bf16 back into the (now dead) x strip.
  // This LDS bounce converts C-layout -> A-layout once per strip. ----
#pragma unroll
  for (int ot = 0; ot < 16; ++ot)
#pragma unroll
    for (int g = 0; g < 8; ++g)
      qs[g + 8 * half][ot * 16 + lrow] = (__bf16)(qacc[ot][g] - qm[g]);

  // Hoist all 8 loop-invariant centered-q A-fragments into VGPRs (64 regs):
  // removes 512 ds_load_b128 per strip from the streaming loop below.
  v16bf afr[8];
#pragma unroll
  for (int kk = 0; kk < 8; ++kk) {
    const int k0 = kk * 32 + half * 16;
    afr[kk] = ld8x2(&qs[lrow][k0], &qs[lrow][k0 + 8]);
  }

  // ---- Phase 5: stream memory slots in chunks of 32: cov -> ssim -> exp ->
  // running sum (ssim is O(1)-bounded: softmax needs no max subtraction) ->
  // att@v accumulation. ----
  v8f oacc[16];
#pragma unroll
  for (int ot = 0; ot < 16; ++ot) { v8f z = {}; oacc[ot] = z; }
  float lacc[8] = {};

  for (int jb = 0; jb < MEMD; jb += 32) {
    v8f s0 = {}, s1 = {};
#pragma unroll
    for (int kk = 0; kk < 8; ++kk) {
      const int kb = kk * 32 + half * 16;
      const v16bf b0 = ld16(kcb + (jb + lrow) * HW + kb);
      s0 = wmma_bf16(afr[kk], b0, s0);
      const v16bf b1 = ld16(kcb + (jb + 16 + lrow) * HW + kb);
      s1 = wmma_bf16(afr[kk], b1, s1);
    }
    const int j0 = jb + lrow, j1 = j0 + 16;
    const float km0 = kmw[j0], kv0 = kvw[j0];
    const float km1 = kmw[j1], kv1 = kvw[j1];
#pragma unroll
    for (int g = 0; g < 8; ++g) {
      const float qmg = qm[g], qvg = qv[g], qm2 = qm[g] * qm[g];
      const float c0 = s0[g] * (1.0f / (HW - 1));
      const float n0 = (2.f * qmg * km0 + C1f) * (2.f * c0 + C2f);
      const float d0 = (qm2 + km0 * km0 + C1f) * (qvg + kv0 + C2f);
      const float p0 = __expf(n0 / (d0 + EPSf));
      const float c1 = s1[g] * (1.0f / (HW - 1));
      const float n1 = (2.f * qmg * km1 + C1f) * (2.f * c1 + C2f);
      const float d1 = (qm2 + km1 * km1 + C1f) * (qvg + kv1 + C2f);
      const float p1 = __expf(n1 / (d1 + EPSf));
      lacc[g] += p0 + p1;
      const int prow = g + 8 * half;           // C-layout row -> LDS p tile
      ps[prow][lrow]      = (__bf16)p0;
      ps[prow][lrow + 16] = (__bf16)p1;
    }
    // att(16x32) @ v(32x256): A from LDS p tile, B from v^T rows (contiguous)
    const v16bf pa = ld8x2(&ps[lrow][half * 16], &ps[lrow][half * 16 + 8]);
#pragma unroll
    for (int ot = 0; ot < 16; ++ot) {
      const v16bf vb = ld16(vTb + (ot * 16 + lrow) * MEMD + jb + half * 16);
      oacc[ot] = wmma_bf16(pa, vb, oacc[ot]);
    }
  }

  // ---- Phase 6: normalize by softmax sum and store ----
  float linv[8];
#pragma unroll
  for (int g = 0; g < 8; ++g) {
    float s = lacc[g];
#pragma unroll
    for (int m = 1; m <= 8; m <<= 1) s += __shfl_xor(s, m, 32);
    linv[g] = 1.0f / s;
  }
#pragma unroll
  for (int ot = 0; ot < 16; ++ot)
#pragma unroll
    for (int g = 0; g < 8; ++g)
      out[(r0 + g + 8 * half) * HW + ot * 16 + lrow] = oacc[ot][g] * linv[g];
}

// ---------------------------------------------------------------------------
extern "C" void kernel_launch(void* const* d_in, const int* in_sizes, int n_in,
                              void* d_out, int out_size, void* d_ws,
                              size_t ws_size, hipStream_t stream) {
  const float* x   = (const float*)d_in[0];  // [8192,8,256]
  const float* mem = (const float*)d_in[1];  // [1024,16,16]
  const float* Wq  = (const float*)d_in[2];  // [256,256]
  const float* Wk  = (const float*)d_in[3];
  const float* Wv  = (const float*)d_in[4];
  float* out = (float*)d_out;                // [8192,8,256]

  char* ws = (char*)d_ws;  // ~1.2 MB total
  __bf16* Wqb = (__bf16*)(ws);                               // 256*256*2
  __bf16* kcb = (__bf16*)(ws + 131072);                      // 1024*256*2
  __bf16* vTb = (__bf16*)(ws + 131072 + 524288);             // 256*1024*2
  float*  kmw = (float*)(ws + 131072 + 524288 + 524288);     // 1024*4
  float*  kvw = (float*)(ws + 131072 + 524288 + 524288 + 4096);

  smm_wconv<<<256, 256, 0, stream>>>(Wq, Wqb);
  smm_prep<<<MEMD, 256, 0, stream>>>(mem, Wk, Wv, kcb, vTb, kmw, kvw);
  smm_main<<<512, 256, 0, stream>>>(x, Wqb, kcb, vTb, kmw, kvw, out);
}
